// SpNet_78219944395503
// MI455X (gfx1250) — matrix operand order repaired and checked by reading, not compiled, not run on hardware
//
#include <hip/hip_runtime.h>
#include <hip/hip_bf16.h>
#include <stdint.h>

// ---------------------------------------------------------------------------
// SpNet on MI455X (gfx1250, wave32).
// All convs are mapped to WMMA f16 GEMMs (v_wmma_f32_16x16x32_f16, f32 acc).
// Activation layout: f16 [point][paddedW][C] with 1 zero-column pad each side,
// so the kw=3 im2col row at output w is a contiguous 3*C span at w*stride*C.
// Conv biases are mathematically irrelevant before batch-stat BN (additive
// per-channel shift cancels in (y-mean)); only final.b is applied.
// GEMM: each wave computes a 16(M) x 32(Co) strip (two co-tiles) so each A
// fragment feeds two independent WMMAs. The K loop is fully unrolled
// (NUMKT is a template parameter, one of {3,6,8,12,24}) so the scheduler
// hoists fragment loads across WMMAs with no register-rotation copies.
// ---------------------------------------------------------------------------

typedef _Float16 v8h  __attribute__((ext_vector_type(8)));
typedef _Float16 v16h __attribute__((ext_vector_type(16)));
typedef float    v8f  __attribute__((ext_vector_type(8)));

#define NPTS 8192

// ---------------- utility -------------------------------------------------
__global__ void fill_zero_u32(uint32_t* __restrict__ p, size_t n) {
  size_t i = (size_t)blockIdx.x * blockDim.x + threadIdx.x;
  if (i < n) p[i] = 0u;
}

// ---------------- weight packing into WMMA B-fragment order ---------------
// B[k][co] with k = kw*Ci + ci ; fragment element for (lane L, elem e) is
// B[kt*32 + (L/16)*16 + e][coTile*16 + (L%16)].
__global__ void prep_b_kernel(const float* __restrict__ w, _Float16* __restrict__ bp,
                              int Ci, int Co, int KW) {
  int K = KW * Ci;
  int idx = blockIdx.x * blockDim.x + threadIdx.x;
  if (idx >= K * Co) return;
  int co = idx % Co;
  int k  = idx / Co;
  int ci = k % Ci;
  int kw = k / Ci;
  float v = w[((size_t)co * Ci + ci) * KW + kw];
  int kt = k >> 5, kin = k & 31;
  int hsel = kin >> 4, e = kin & 15;
  int lane = hsel * 16 + (co & 15);
  int t = co >> 4;
  int numKt = K >> 5;
  bp[(((size_t)t * numKt + kt) * 32 + lane) * 16 + e] = (_Float16)v;
}

// ---------------- WMMA GEMM -----------------------------------------------
// Wave computes 16 rows x 32 cols (two 16x16 co-tiles). grid:
//   x = M/(16*blockDim.y), y = Co/32, z = W (output width position).
// act: f16 padded activations, row stride rowStrideH halfs.
// For output position w the K-span begins at colBaseH + w*colPerWH.
// y layout: [n][w][co] f32. K = NUMKT*32 (compile-time).
template <int NUMKT>
__global__ __launch_bounds__(128) void gemm_wmma_kernel(
    const _Float16* __restrict__ act, int rowStrideH, int colBaseH, int colPerWH,
    const _Float16* __restrict__ bpack, float* __restrict__ y,
    int Co, int W) {
  int lane = threadIdx.x;        // 0..31
  int hl   = lane >> 4;          // wave half
  int mrow = lane & 15;
  int mBase = (blockIdx.x * blockDim.y + threadIdx.y) * 16;
  int coT2 = blockIdx.y;         // pair of co tiles
  int w   = blockIdx.z;

  const _Float16* Arow = act + (size_t)(mBase + mrow) * rowStrideH
                             + colBaseH + (size_t)w * colPerWH + hl * 16;
  const _Float16* Bp = bpack + (((size_t)(2 * coT2) * NUMKT) * 32 + lane) * 16;
  const _Float16* Bq = bpack + (((size_t)(2 * coT2 + 1) * NUMKT) * 32 + lane) * 16;

  v8f c0 = {0.f, 0.f, 0.f, 0.f, 0.f, 0.f, 0.f, 0.f};
  v8f c1 = {0.f, 0.f, 0.f, 0.f, 0.f, 0.f, 0.f, 0.f};

#pragma unroll
  for (int kt = 0; kt < NUMKT; ++kt) {
    v8h alo = *(const v8h*)(Arow + kt * 32);
    v8h ahi = *(const v8h*)(Arow + kt * 32 + 8);
    v16h b0 = *(const v16h*)(Bp + (size_t)kt * 512);
    v16h b1 = *(const v16h*)(Bq + (size_t)kt * 512);
    v16h a = __builtin_shufflevector(alo, ahi,
        0, 1, 2, 3, 4, 5, 6, 7, 8, 9, 10, 11, 12, 13, 14, 15);
    c0 = __builtin_amdgcn_wmma_f32_16x16x32_f16(
        false, a, false, b0, (short)0, c0, false, false);
    c1 = __builtin_amdgcn_wmma_f32_16x16x32_f16(
        false, a, false, b1, (short)0, c1, false, false);
  }

  // C/D layout: lane half -> rows v + hl*8, column = lane%16
  size_t outRow = (size_t)Co * W;
  float* yp0 = y + (size_t)w * Co + (2 * coT2) * 16 + mrow;
  float* yp1 = yp0 + 16;
#pragma unroll
  for (int v = 0; v < 8; ++v) {
    size_t r = (size_t)(mBase + hl * 8 + v) * outRow;
    yp0[r] = c0[v];
    yp1[r] = c1[v];
  }
}

// ---------------- batch-stat reduction ------------------------------------
__global__ __launch_bounds__(256) void stats_kernel(
    const float* __restrict__ y, int rowsPerBlock, int C, float* __restrict__ stats) {
  __shared__ float shs[256];
  __shared__ float shq[256];
  int numSub = 256 / C;                 // C in {32,64,128,256}
  int c = threadIdx.x % C;
  int sub = threadIdx.x / C;
  int r0 = blockIdx.x * rowsPerBlock;
  float s = 0.f, q = 0.f;
  for (int r = sub; r < rowsPerBlock; r += numSub) {
    float v = y[(size_t)(r0 + r) * C + c];
    s += v;
    q += v * v;
  }
  shs[threadIdx.x] = s;
  shq[threadIdx.x] = q;
  __syncthreads();
  for (int off = numSub >> 1; off > 0; off >>= 1) {
    if (sub < off) {
      shs[threadIdx.x] += shs[threadIdx.x + off * C];
      shq[threadIdx.x] += shq[threadIdx.x + off * C];
    }
    __syncthreads();
  }
  if (sub == 0) {
    atomicAdd(&stats[c], shs[threadIdx.x]);
    atomicAdd(&stats[C + c], shq[threadIdx.x]);
  }
}

__global__ void bn_finalize_kernel(const float* __restrict__ stats,
                                   const float* __restrict__ g,
                                   const float* __restrict__ beta,
                                   float cnt, int C, float* __restrict__ ss) {
  int c = blockIdx.x * blockDim.x + threadIdx.x;
  if (c >= C) return;
  float mean = stats[c] / cnt;
  float var = fmaxf(stats[C + c] / cnt - mean * mean, 0.f);
  float sc = g[c] * rsqrtf(var + 1e-5f);
  ss[c] = sc;
  ss[C + c] = beta[c] - mean * sc;
}

// ---------------- fused BN (+residual)(+relu) + layout stores -------------
__global__ void bn_apply_kernel(const float* __restrict__ y, const float* __restrict__ ss,
                                const float* __restrict__ identity, int doRelu,
                                int rows, int C, int W,
                                _Float16* __restrict__ actOut, int actRowStrideH,
                                float* __restrict__ resOut) {
  int idx = blockIdx.x * blockDim.x + threadIdx.x;
  if (idx >= rows * C) return;
  int c = idx % C;
  int r = idx / C;
  float v = y[idx] * ss[c] + ss[C + c];
  if (identity) v += identity[idx];
  if (doRelu) v = fmaxf(v, 0.f);
  if (resOut) resOut[idx] = v;
  if (actOut) {
    int w = r % W;
    int n = r / W;
    actOut[(size_t)n * actRowStrideH + (size_t)(w + 1) * C + c] = (_Float16)v;
  }
}

// ---------------- stem: 6->32 linear (bias cancels in BN) -----------------
__global__ void stem_mm_kernel(const float* __restrict__ x, const float* __restrict__ w,
                               float* __restrict__ y, int N) {
  int idx = blockIdx.x * blockDim.x + threadIdx.x;
  if (idx >= N * 32) return;
  int n = idx >> 5, co = idx & 31;
  const float* xr = x + (size_t)n * 6;
  const float* wr = w + (size_t)co * 6;
  float s = 0.f;
#pragma unroll
  for (int k = 0; k < 6; ++k) s = fmaf(wr[k], xr[k], s);
  y[idx] = s;
}

// ---------------- kNN: top-8 smallest distances (self included) -----------
__global__ __launch_bounds__(256) void knn_kernel(const float* __restrict__ x,
                                                  int* __restrict__ nbr, int N) {
  __shared__ float sx[256 * 7];
  int n = blockIdx.x * 256 + threadIdx.x;
  float xr[6];
#pragma unroll
  for (int k = 0; k < 6; ++k) xr[k] = x[(size_t)n * 6 + k];
  float sq = 0.f;
#pragma unroll
  for (int k = 0; k < 6; ++k) sq = fmaf(xr[k], xr[k], sq);
  float bd[8];
  int bi[8];
#pragma unroll
  for (int i = 0; i < 8; ++i) { bd[i] = 3.4e38f; bi[i] = 0; }
  for (int t = 0; t < N; t += 256) {
    __syncthreads();
    {
      int j = t + threadIdx.x;
      float s = 0.f;
#pragma unroll
      for (int k = 0; k < 6; ++k) {
        float v = x[(size_t)j * 6 + k];
        sx[threadIdx.x * 7 + k] = v;
        s = fmaf(v, v, s);
      }
      sx[threadIdx.x * 7 + 6] = s;
    }
    __syncthreads();
    for (int jj = 0; jj < 256; ++jj) {
      const float* p = &sx[jj * 7];
      float dot = 0.f;
#pragma unroll
      for (int k = 0; k < 6; ++k) dot = fmaf(xr[k], p[k], dot);
      float d = fmaxf(sq + p[6] - 2.f * dot, 0.f);
      if (d < bd[7]) {
        int j = t + jj;
        int q = 7;
        while (q > 0 && bd[q - 1] > d) { bd[q] = bd[q - 1]; bi[q] = bi[q - 1]; --q; }
        bd[q] = d;
        bi[q] = j;
      }
    }
  }
#pragma unroll
  for (int i = 0; i < 8; ++i) nbr[(size_t)n * 8 + i] = bi[i];
}

// ---------------- neighbor gather -> first block activation ---------------
__global__ void gather_kernel(const float* __restrict__ feat, const int* __restrict__ nbr,
                              _Float16* __restrict__ act0, float* __restrict__ res0, int N) {
  int idx = blockIdx.x * blockDim.x + threadIdx.x;
  if (idx >= N * 8 * 32) return;
  int c = idx & 31;
  int j = (idx >> 5) & 7;
  int n = idx >> 8;
  int src = nbr[(size_t)n * 8 + j];
  float v = feat[(size_t)src * 32 + c];
  act0[(size_t)n * 320 + (size_t)(j + 1) * 32 + c] = (_Float16)v;  // padded [n][10][32]
  res0[idx] = v;                                                   // [n][8][32]
}

// ---------------- head: 256->1 dot + bias, tanh/relu ----------------------
__global__ __launch_bounds__(256) void final_kernel(const float* __restrict__ a,
                                                    const float* __restrict__ wf,
                                                    const float* __restrict__ bf,
                                                    float* __restrict__ out, int N) {
  int lane = threadIdx.x;
  int n = blockIdx.x * 8 + threadIdx.y;
  float s = 0.f;
  for (int i = lane; i < 256; i += 32) s = fmaf(a[(size_t)n * 256 + i], wf[i], s);
#pragma unroll
  for (int off = 16; off > 0; off >>= 1) s += __shfl_xor(s, off, 32);
  if (lane == 0) {
    float o = s + bf[0];
    out[n] = o;
    out[N + n] = fmaxf(tanhf(o), 0.f);
  }
}

// ===========================================================================
extern "C" void kernel_launch(void* const* d_in, const int* in_sizes, int n_in,
                              void* d_out, int out_size, void* d_ws, size_t ws_size,
                              hipStream_t stream) {
  (void)in_sizes; (void)out_size; (void)ws_size;
  const int N = NPTS;
  if (n_in < 87) return;

  // ---- input index map: JAX sorted-key pytree flatten ----
  // params.blocks[b]: bn1{beta,g} bn2{beta,g} [bnr{beta,g}] l1{b,w} l2{b,w} [r{b,w}]
  // then params.bn1{beta,g} bn2{beta,g} conv1{b,w} conv2{b,w} final{b,w}, then x.
  const int pre[8]     = {0, 0, 1, 0, 1, 0, 1, 0};
  const int CiArr[8]   = {32, 32, 32, 64, 64, 128, 128, 256};
  const int CoArr[8]   = {32, 32, 64, 64, 128, 128, 256, 256};
  const int stArr[8]   = {1, 1, 2, 1, 2, 1, 2, 1};
  int i_bn1b[8], i_bn1g[8], i_bn2b[8], i_bn2g[8], i_bnrb[8], i_bnrg[8];
  int i_l1w[8], i_l2w[8], i_rw[8];
  int base = 0;
  for (int b = 0; b < 8; ++b) {
    i_bn1b[b] = base + 0; i_bn1g[b] = base + 1;
    i_bn2b[b] = base + 2; i_bn2g[b] = base + 3;
    if (pre[b]) {
      i_bnrb[b] = base + 4; i_bnrg[b] = base + 5;
      i_l1w[b] = base + 7; i_l2w[b] = base + 9; i_rw[b] = base + 11;
      base += 12;
    } else {
      i_bnrb[b] = -1; i_bnrg[b] = -1;
      i_l1w[b] = base + 5; i_l2w[b] = base + 7; i_rw[b] = -1;
      base += 8;
    }
  }
  const int iBN1b = base + 0, iBN1g = base + 1;
  const int iBN2b = base + 2, iBN2g = base + 3;
  const int iC1w = base + 5;                    // conv1.{b=+4,w=+5}
  const int iC2w = base + 7;                    // conv2.{b=+6,w=+7}
  const int iFinB = base + 8, iFinW = base + 9; // final.{b,w}
  const int iX = base + 10;

  // ---- workspace layout ----
  char* wsb = (char*)d_ws;
  size_t off = 0;
  auto carve = [&](size_t bytes) { size_t o = off; off += (bytes + 255) & ~(size_t)255; return o; };
  _Float16* ACTA  = (_Float16*)(wsb + carve((size_t)N * 768 * 2)); // padded f16 act (max [N][3][256])
  _Float16* ACTB  = (_Float16*)(wsb + carve((size_t)N * 768 * 2));
  float* Y1   = (float*)(wsb + carve((size_t)N * 256 * 4));        // pre-BN conv outputs (W*C <= 256)
  float* Y2   = (float*)(wsb + carve((size_t)N * 256 * 4));
  float* YR   = (float*)(wsb + carve((size_t)N * 256 * 4));
  float* RES0 = (float*)(wsb + carve((size_t)N * 256 * 4));        // residual ping
  float* RES1 = (float*)(wsb + carve((size_t)N * 256 * 4));        // residual pong
  float* IDN  = (float*)(wsb + carve((size_t)N * 256 * 4));        // projected identity
  float* FEAT = (float*)(wsb + carve((size_t)N * 32 * 4));
  float* YST  = (float*)(wsb + carve((size_t)N * 32 * 4));
  int*   NBR  = (int*)(wsb + carve((size_t)N * 8 * 4));
  float* STATS = (float*)(wsb + carve(2048));
  float* SS    = (float*)(wsb + carve(2048));
  _Float16* BP = (_Float16*)(wsb + carve((size_t)4 << 20));

  auto F = [&](int i) { return (const float*)d_in[i]; };
  auto zero = [&](void* p, size_t bytes) {
    size_t n32 = bytes / 4;
    fill_zero_u32<<<(unsigned)((n32 + 255) / 256), 256, 0, stream>>>((uint32_t*)p, n32);
  };
  auto runStats = [&](const float* y, int rows, int C) {
    zero(STATS, (size_t)2 * C * 4);
    stats_kernel<<<64, 256, 0, stream>>>(y, rows / 64, C, STATS);
  };
  auto runFin = [&](int gi, int bi, int rows, int C) {
    bn_finalize_kernel<<<(C + 63) / 64, 64, 0, stream>>>(STATS, F(gi), F(bi), (float)rows, C, SS);
  };
  auto runGemm = [&](const _Float16* act, int rsH, int colBase, int colPerW,
                     const _Float16* bp, float* y, int K, int Co, int W) {
    dim3 blk(32, 4, 1), grd(N / 64, Co / 32, W);
    switch (K >> 5) {
      case 3:
        gemm_wmma_kernel<3><<<grd, blk, 0, stream>>>(act, rsH, colBase, colPerW, bp, y, Co, W);
        break;
      case 6:
        gemm_wmma_kernel<6><<<grd, blk, 0, stream>>>(act, rsH, colBase, colPerW, bp, y, Co, W);
        break;
      case 8:
        gemm_wmma_kernel<8><<<grd, blk, 0, stream>>>(act, rsH, colBase, colPerW, bp, y, Co, W);
        break;
      case 12:
        gemm_wmma_kernel<12><<<grd, blk, 0, stream>>>(act, rsH, colBase, colPerW, bp, y, Co, W);
        break;
      case 24:
        gemm_wmma_kernel<24><<<grd, blk, 0, stream>>>(act, rsH, colBase, colPerW, bp, y, Co, W);
        break;
      default:
        break;  // not used by this network
    }
  };
  auto runApply = [&](const float* y, const float* id, int relu, int rows, int C, int W,
                      _Float16* actOut, int rsH, float* resOut) {
    int tot = rows * C;
    bn_apply_kernel<<<(tot + 255) / 256, 256, 0, stream>>>(y, SS, id, relu, rows, C, W,
                                                           actOut, rsH, resOut);
  };

  // ---- 0. pack all conv weights into WMMA B-fragment order ----
  size_t bpL1[8], bpL2[8], bpR[8], bpC2, cur = 0;
  for (int b = 0; b < 8; ++b) {
    int Ci = CiArr[b], Co = CoArr[b];
    bpL1[b] = cur;
    prep_b_kernel<<<(3 * Ci * Co + 255) / 256, 256, 0, stream>>>(F(i_l1w[b]), BP + cur, Ci, Co, 3);
    cur += (size_t)3 * Ci * Co;
    bpL2[b] = cur;
    prep_b_kernel<<<(3 * Co * Co + 255) / 256, 256, 0, stream>>>(F(i_l2w[b]), BP + cur, Co, Co, 3);
    cur += (size_t)3 * Co * Co;
    if (pre[b]) {
      bpR[b] = cur;
      prep_b_kernel<<<(3 * Ci * Co + 255) / 256, 256, 0, stream>>>(F(i_rw[b]), BP + cur, Ci, Co, 3);
      cur += (size_t)3 * Ci * Co;
    } else bpR[b] = 0;
  }
  bpC2 = cur;
  prep_b_kernel<<<(256 * 256 + 255) / 256, 256, 0, stream>>>(F(iC2w), BP + cur, 256, 256, 1);

  // ---- 1. stem: linear 6->32, BN(batch stats), ReLU -> FEAT ----
  stem_mm_kernel<<<(N * 32) / 256, 256, 0, stream>>>(F(iX), F(iC1w), YST, N);
  runStats(YST, N, 32);
  runFin(iBN1g, iBN1b, N, 32);
  runApply(YST, nullptr, 1, N, 32, 1, nullptr, 0, FEAT);

  // ---- 2. kNN top-8 + gather into padded [N][10][32] f16 + residual f32 ----
  knn_kernel<<<N / 256, 256, 0, stream>>>(F(iX), NBR, N);
  zero(ACTA, (size_t)N * 320 * 2);
  gather_kernel<<<(N * 8 * 32) / 256, 256, 0, stream>>>(FEAT, NBR, ACTA, RES0, N);

  // ---- 3. residual blocks ----
  int Wi = 8, Ci = 32;
  float* resCur = RES0;
  float* resNext = RES1;
  for (int b = 0; b < 8; ++b) {
    int Co = CoArr[b], st = stArr[b], Wo = Wi / st;
    int rsIn = (Wi + 2) * Ci;
    int rsMid = (Wo + 2) * Co;
    int rows = N * Wo;

    // l1: conv -> BN -> ReLU -> padded f16 (ACTB)
    runGemm(ACTA, rsIn, 0, st * Ci, BP + bpL1[b], Y1, 3 * Ci, Co, Wo);
    runStats(Y1, rows, Co);
    runFin(i_bn1g[b], i_bn1b[b], rows, Co);
    zero(ACTB, (size_t)N * rsMid * 2);
    runApply(Y1, nullptr, 1, rows, Co, Wo, ACTB, rsMid, nullptr);

    // l2: conv (pre-BN output in Y2)
    runGemm(ACTB, rsMid, 0, Co, BP + bpL2[b], Y2, 3 * Co, Co, Wo);

    // identity path
    const float* idPtr;
    if (pre[b]) {
      runGemm(ACTA, rsIn, 0, st * Ci, BP + bpR[b], YR, 3 * Ci, Co, Wo);
      runStats(YR, rows, Co);
      runFin(i_bnrg[b], i_bnrb[b], rows, Co);
      runApply(YR, nullptr, 0, rows, Co, Wo, nullptr, 0, IDN);
      idPtr = IDN;
    } else {
      idPtr = resCur;
    }

    // out = relu(bn(l2) + identity) -> next padded act (ACTA) + next residual
    runStats(Y2, rows, Co);
    runFin(i_bn2g[b], i_bn2b[b], rows, Co);
    zero(ACTA, (size_t)N * rsMid * 2);  // safe: all reads of ACTA done above
    runApply(Y2, idPtr, 1, rows, Co, Wo, ACTA, rsMid, resNext);

    float* t = resCur; resCur = resNext; resNext = t;
    Wi = Wo;
    Ci = Co;
  }

  // ---- 4. conv2 (1x1, 256->256) + BN + ReLU ----
  runGemm(ACTA, 3 * 256, 256, 0, BP + bpC2, Y1, 256, 256, 1);
  runStats(Y1, N, 256);
  runFin(iBN2g, iBN2b, N, 256);
  runApply(Y1, nullptr, 1, N, 256, 1, nullptr, 0, resNext);

  // ---- 5. head: out = final(a2); w = relu(tanh(out)) ----
  final_kernel<<<dim3(N / 8), dim3(32, 8), 0, stream>>>(resNext, F(iFinW), F(iFinB),
                                                        (float*)d_out, N);
}
